// RewardModel_27384711480162
// MI455X (gfx1250) — matrix operand order repaired and checked by reading, not compiled, not run on hardware
//
#include <hip/hip_runtime.h>
#include <hip/hip_bf16.h>
#include <math.h>

#define N_NODES   8192
#define N_EDGES   131072
#define N_GRAPHS  8
#define NODE_FDIM 64
#define NS        16
#define DIST_DIM  32

// RBF params: mu = linspace(0,10,32)
#define DMU  (10.0f / 31.0f)
#define RBFC (-0.5f / (DMU * DMU))

// coupling constants from reference
#define C_L0_00 0.25f
#define C_L0_01 0.25f
#define C1_P1 0.17677669529663687f   // sqrt(1/32)
#define C1_P2 0.14433756729740643f   // 0.25/sqrt(3)
#define C1_P3 0.14433756729740643f
#define C1_P4 0.10206207261596575f   // sqrt(1/32)/sqrt(3)
#define C1_P5 0.17677669529663687f   // sqrt(3/16)/sqrt(6)
#define C1_P6 0.11180339887498949f   // 0.25/sqrt(5)
#define SQRT3 1.7320508075688772f
#define SQRT7P5 2.7386127875258306f
#define S2C 0.7071067811865476f      // 1/sqrt(2)
#define S6C 0.4082482904638631f      // 1/sqrt(6)

typedef __bf16 bf16_t;
typedef bf16_t v16bf __attribute__((ext_vector_type(16)));
typedef float  v8f   __attribute__((ext_vector_type(8)));

__device__ __forceinline__ v8f wmma_bf16(v16bf a, v16bf b, v8f c) {
  // D = A(16x32 bf16) * B(32x16 bf16) + C(16x16 f32)
  return __builtin_amdgcn_wmma_f32_16x16x32_bf16(false, a, false, b, (short)0, c,
                                                 false, false);
}

// ---------------------------------------------------------------------------
// Packed A tile: the 16x64 bf16 edge-feature / hidden tile is stored in LDS in
// WMMA A-fragment order, so each lane's fragment is 16 contiguous bf16 (32 B).
// A-layout (16-bit, 16x32): lanes 0-15 & 16-31 both hold rows M=0..15;
//   element i<8 : K = k0 + 8*half + i
//   element i>=8: K = k0 + 16 + 8*half + (i-8)
// afrag_idx maps (row m, col k) -> linear index in that packed order.
__device__ __forceinline__ int afrag_idx(int m, int k) {
  const int chunk = k >> 5;             // which 16x32 A operand
  const int kc = k & 31;
  const int hf = (kc >> 3) & 1;         // lane half that owns this k
  const int i = (kc & 7) + ((kc & 16) ? 8 : 0);
  return ((chunk * 2 + hf) * 16 + m) * 16 + i;
}

__device__ __forceinline__ v16bf load_a_packed(const bf16_t* tileB, int l16, int half,
                                               int chunk, bool zero_hi) {
  const bf16_t* p = tileB + ((chunk * 2 + half) * 16 + l16) * 16;
  v16bf a = {};
#pragma unroll
  for (int i = 0; i < 8; ++i) a[i] = p[i];
  if (!zero_hi) {
#pragma unroll
    for (int i = 8; i < 16; ++i) a[i] = p[i];
  }
  return a;
}

// B fragment (32x16 slice of weights) from bf16 LDS stored transposed [N][Kld]:
// element i<8: K = k0 + 8*half + i; i>=8: K = k0 + 16 + 8*half + (i-8); N = n0 + (lane&15).
// Our K tails are whole 8-element groups (48-wide K), so skip loads
// per group instead of per-element masking.
__device__ __forceinline__ v16bf load_b_frag(const bf16_t* wT, int Kld, int n0,
                                             int l16, int half, int k0, int kmax) {
  v16bf b = {};
  const int kb = k0 + 8 * half;
  const bf16_t* p = wT + (n0 + l16) * Kld + kb;
  if (kb < kmax) {
#pragma unroll
    for (int i = 0; i < 8; ++i) b[i] = p[i];
  }
  if (kb + 16 < kmax) {
#pragma unroll
    for (int i = 8; i < 16; ++i) b[i] = p[i + 8];
  }
  return b;
}

// ---------------------------------------------------------------- embed (h0)
__global__ void embed_kernel(const float* __restrict__ x,
                             const float* __restrict__ w1, const float* __restrict__ b1,
                             const float* __restrict__ w2, const float* __restrict__ b2,
                             float* __restrict__ h0) {
  const int n = blockIdx.x * blockDim.x + threadIdx.x;
  if (n >= N_NODES) return;
  float hid[16];
#pragma unroll
  for (int j = 0; j < 16; ++j) hid[j] = b1[j];
  for (int k = 0; k < 64; ++k) {
    const float xv = x[n * 64 + k];
#pragma unroll
    for (int j = 0; j < 16; ++j) hid[j] += xv * w1[k * 16 + j];
  }
#pragma unroll
  for (int j = 0; j < 16; ++j) hid[j] = fmaxf(hid[j], 0.0f);
  float o[16];
#pragma unroll
  for (int j = 0; j < 16; ++j) o[j] = b2[j];
#pragma unroll
  for (int k = 0; k < 16; ++k)
#pragma unroll
    for (int j = 0; j < 16; ++j) o[j] += hid[k] * w2[k * 16 + j];
#pragma unroll
  for (int j = 0; j < 16; ++j) h0[n * 16 + j] = o[j];
}

// ---------------------------------------------------------------- degree
__global__ void deg_kernel(const int* __restrict__ dst, float* __restrict__ deg) {
  const int e = blockIdx.x * blockDim.x + threadIdx.x;
  if (e < N_EDGES) atomicAdd(&deg[dst[e]], 1.0f);
}

// ---------------------------------------------------------------- conv0 (fused)
// WG = 256 threads = 8 waves; each wave owns a 16-edge tile (WMMA M dim).
__global__ void conv0_kernel(const float* __restrict__ pos,
                             const int* __restrict__ src, const int* __restrict__ dst,
                             const float* __restrict__ h0,
                             const float* __restrict__ w1, const float* __restrict__ b1,
                             const float* __restrict__ w2, const float* __restrict__ b2,
                             float* __restrict__ up0) {
  extern __shared__ char smem[];
  bf16_t* w1T = (bf16_t*)smem;                                  // [48][64]
  float*  b1s = (float*)(smem + 48 * 64 * 2);                   // 48
  bf16_t* w2T = (bf16_t*)(smem + 48 * 64 * 2 + 48 * 4);         // [512][48]
  float*  b2s = (float*)(smem + 48 * 64 * 2 + 48 * 4 + 512 * 48 * 2);
  char* wbase = smem + 48 * 64 * 2 + 48 * 4 + 512 * 48 * 2 + 512 * 4;  // 57536

  const int tid = threadIdx.x, lane = tid & 31, wv = tid >> 5;
  const int l16 = lane & 15, half = lane >> 4;
  const int WB = 2048 + 1024 + 192 + 64;       // 3328 B per wave
  char* wvb = wbase + wv * WB;
  bf16_t* tileB = (bf16_t*)wvb;                // packed 16x64 bf16 (ef, then hmid)
  float*  xsl   = (float*)(wvb + 2048);        // [16][16] xs = h0[src]
  float*  sh1l  = xsl + 256;                   // [16][3]
  int*    dstl  = (int*)(sh1l + 48);           // [16]

  // stage weights transposed into LDS as bf16
  for (int i = tid; i < 48 * 64; i += 256) {
    const int n = i >> 6, k = i & 63;
    w1T[i] = (bf16_t)w1[k * 48 + n];
  }
  for (int i = tid; i < 512 * 48; i += 256) {
    const int n = i / 48, k = i - n * 48;
    w2T[i] = (bf16_t)w2[k * 512 + n];
  }
  for (int i = tid; i < 48; i += 256)  b1s[i] = b1[i];
  for (int i = tid; i < 512; i += 256) b2s[i] = b2[i];
  __syncthreads();

  const int e = (blockIdx.x * 8 + wv) * 16 + l16;
  {
    const int s = src[e], d = dst[e];
    const float dx = pos[d * 3 + 0] - pos[s * 3 + 0];
    const float dy = pos[d * 3 + 1] - pos[s * 3 + 1];
    const float dz = pos[d * 3 + 2] - pos[s * 3 + 2];
    const float rr = sqrtf(dx * dx + dy * dy + dz * dz);
    const float inv = 1.0f / (rr + 1e-8f);
    const float rx = dx * inv, ry = dy * inv, rz = dz * inv;
#pragma unroll
    for (int k = 0; k < 16; ++k) {  // edge_attr cols [0..31], split by half
      const int kk = 16 * half + k;
      const float t = rr - DMU * (float)kk;
      tileB[afrag_idx(l16, kk)] = (bf16_t)__expf(RBFC * t * t);
    }
    if (half == 0) {
#pragma unroll
      for (int k = 0; k < 16; ++k)
        tileB[afrag_idx(l16, 32 + k)] = (bf16_t)h0[d * 16 + k];
      dstl[l16] = d;
      sh1l[l16 * 3 + 0] = SQRT3 * rx;
      sh1l[l16 * 3 + 1] = SQRT3 * ry;
      sh1l[l16 * 3 + 2] = SQRT3 * rz;
    } else {
#pragma unroll
      for (int k = 0; k < 16; ++k) {
        const float v = h0[s * 16 + k];
        tileB[afrag_idx(l16, 48 + k)] = (bf16_t)v;  // ef cols 48..63 = h0[src]
        xsl[l16 * 16 + k] = v;
      }
    }
  }

  // GEMM1: hmid = relu(ef @ w1 + b1), 16x64 @ 64x48
  {
    const v16bf a0 = load_a_packed(tileB, l16, half, 0, false);
    const v16bf a1 = load_a_packed(tileB, l16, half, 1, false);
    float hm[3][8];
#pragma unroll
    for (int nb = 0; nb < 3; ++nb) {
      v8f c = {};
      c = wmma_bf16(a0, load_b_frag(w1T, 64, nb * 16, l16, half, 0, 64), c);
      c = wmma_bf16(a1, load_b_frag(w1T, 64, nb * 16, l16, half, 32, 64), c);
      const float bias = b1s[nb * 16 + l16];
#pragma unroll
      for (int r = 0; r < 8; ++r) hm[nb][r] = fmaxf(c[r] + bias, 0.0f);
    }
#pragma unroll
    for (int nb = 0; nb < 3; ++nb)
#pragma unroll
      for (int r = 0; r < 8; ++r)
        tileB[afrag_idx(r + 8 * half, nb * 16 + l16)] = (bf16_t)hm[nb][r];
  }

  // GEMM2 fused with contraction: w = hmid @ w2 + b2;
  // block u of W0 -> o0[e,w] += xs[e,u]*W0[u,w]; of W1 -> s1[e,w].
  v8f accO0 = {}, accS1 = {};
  {
    const v16bf ha0 = load_a_packed(tileB, l16, half, 0, false);
    const v16bf ha1 = load_a_packed(tileB, l16, half, 1, true);  // K 48..63 zero
#pragma unroll 4
    for (int u = 0; u < 16; ++u) {                               // W0
      v8f c = {};
      c = wmma_bf16(ha0, load_b_frag(w2T, 48, u * 16, l16, half, 0, 48), c);
      c = wmma_bf16(ha1, load_b_frag(w2T, 48, u * 16, l16, half, 32, 48), c);
      const float bias = b2s[u * 16 + l16];
#pragma unroll
      for (int r = 0; r < 8; ++r)
        accO0[r] += xsl[(r + 8 * half) * 16 + u] * (c[r] + bias);
    }
#pragma unroll 4
    for (int u = 0; u < 16; ++u) {                               // W1
      const int n0 = 256 + u * 16;
      v8f c = {};
      c = wmma_bf16(ha0, load_b_frag(w2T, 48, n0, l16, half, 0, 48), c);
      c = wmma_bf16(ha1, load_b_frag(w2T, 48, n0, l16, half, 32, 48), c);
      const float bias = b2s[n0 + l16];
#pragma unroll
      for (int r = 0; r < 8; ++r)
        accS1[r] += xsl[(r + 8 * half) * 16 + u] * (c[r] + bias);
    }
  }

  // scatter-add: out cols [0..15]=o0, [16 + 3w + j]=0.25*s1*sh1[j]
#pragma unroll
  for (int r = 0; r < 8; ++r) {
    const int er = r + 8 * half;
    const int d = dstl[er];
    atomicAdd(&up0[d * 64 + l16], C_L0_00 * accO0[r]);
    const float s1 = C_L0_01 * accS1[r];
#pragma unroll
    for (int j = 0; j < 3; ++j)
      atomicAdd(&up0[d * 64 + 16 + 3 * l16 + j], s1 * sh1l[er * 3 + j]);
  }
}

// ---------------------------------------------------------------- node update
__global__ void node_update_kernel(const float* __restrict__ hin, int win,
                                   const float* __restrict__ up, int wup,
                                   const float* __restrict__ deg,
                                   float* __restrict__ hout) {
  const int i = blockIdx.x * blockDim.x + threadIdx.x;
  if (i >= N_NODES * wup) return;
  const int n = i / wup, c = i - n * wup;
  const float base = (c < win) ? hin[n * win + c] : 0.0f;
  hout[i] = base + up[i] / fmaxf(deg[n], 1.0f);
}

// ---------------------------------------------------------------- conv1 (fused)
__global__ void conv1_kernel(const float* __restrict__ pos,
                             const int* __restrict__ src, const int* __restrict__ dst,
                             const float* __restrict__ h1,
                             const float* __restrict__ w1, const float* __restrict__ b1,
                             const float* __restrict__ w2, const float* __restrict__ b2,
                             float* __restrict__ up1) {
  extern __shared__ char smem[];
  bf16_t* w1T = (bf16_t*)smem;                          // [48][64]
  float*  b1s = (float*)(smem + 6144);                  // 48
  bf16_t* w2T = (bf16_t*)(smem + 6144 + 192);           // [1536][48]
  float*  b2s = (float*)(smem + 6144 + 192 + 147456);   // 1536
  char* wbase = smem + 6144 + 192 + 147456 + 6144;      // 159936

  const int tid = threadIdx.x, lane = tid & 31, wv = tid >> 5;
  const int l16 = lane & 15, half = lane >> 4;
  const int WB = 2048 + (256 + 768 + 256 + 768 + 768 + 48 + 16) * 4;  // 13568 B
  char* wvb = wbase + wv * WB;
  bf16_t* tileB = (bf16_t*)wvb;            // packed 16x64 bf16
  float*  xsl   = (float*)(wvb + 2048);    // [16][16]    xs
  float*  xvl   = xsl + 256;               // [16][16][3] xv
  float*  d3l   = xvl + 768;               // [16][16]    xv.sh1
  float*  crl   = d3l + 256;               // [16][16][3] xv x sh1
  float*  xvm2l = crl + 768;               // [16][16][3] xv.m2
  float*  sh1l  = xvm2l + 768;             // [16][3]
  int*    dstl  = (int*)(sh1l + 48);       // [16]

  for (int i = tid; i < 48 * 64; i += 256) {
    const int n = i >> 6, k = i & 63;
    w1T[i] = (bf16_t)w1[k * 48 + n];
  }
  for (int i = tid; i < 1536 * 48; i += 256) {
    const int n = i / 48, k = i - n * 48;
    w2T[i] = (bf16_t)w2[k * 1536 + n];
  }
  for (int i = tid; i < 48; i += 256)   b1s[i] = b1[i];
  for (int i = tid; i < 1536; i += 256) b2s[i] = b2[i];
  __syncthreads();

  const int e = (blockIdx.x * 8 + wv) * 16 + l16;
  {
    const int s = src[e], d = dst[e];
    const float dx = pos[d * 3 + 0] - pos[s * 3 + 0];
    const float dy = pos[d * 3 + 1] - pos[s * 3 + 1];
    const float dz = pos[d * 3 + 2] - pos[s * 3 + 2];
    const float rr = sqrtf(dx * dx + dy * dy + dz * dz);
    const float inv = 1.0f / (rr + 1e-8f);
    const float rx = dx * inv, ry = dy * inv, rz = dz * inv;
    const float s1x = SQRT3 * rx, s1y = SQRT3 * ry, s1z = SQRT3 * rz;
    // sh2 and m2 = sum_k B[k]*sh2[k]
    const float sh2_0 = SQRT7P5 * 1.41421356f * rx * ry;
    const float sh2_1 = SQRT7P5 * 1.41421356f * ry * rz;
    const float sh2_2 = SQRT7P5 * S6C * (2.0f * rz * rz - rx * rx - ry * ry);
    const float sh2_3 = SQRT7P5 * 1.41421356f * rx * rz;
    const float sh2_4 = SQRT7P5 * S2C * (rx * rx - ry * ry);
    float m2[3][3];
    m2[0][0] = -S6C * sh2_2 + S2C * sh2_4;
    m2[0][1] =  S2C * sh2_0;
    m2[0][2] =  S2C * sh2_3;
    m2[1][0] =  S2C * sh2_0;
    m2[1][1] = -S6C * sh2_2 - S2C * sh2_4;
    m2[1][2] =  S2C * sh2_1;
    m2[2][0] =  S2C * sh2_3;
    m2[2][1] =  S2C * sh2_1;
    m2[2][2] =  2.0f * S6C * sh2_2;

#pragma unroll
    for (int k = 0; k < 16; ++k) {  // edge_attr
      const int kk = 16 * half + k;
      const float t = rr - DMU * (float)kk;
      tileB[afrag_idx(l16, kk)] = (bf16_t)__expf(RBFC * t * t);
    }
    if (half == 0) {
#pragma unroll
      for (int k = 0; k < 16; ++k)
        tileB[afrag_idx(l16, 32 + k)] = (bf16_t)h1[d * 64 + k];
      dstl[l16] = d;
      sh1l[l16 * 3 + 0] = s1x; sh1l[l16 * 3 + 1] = s1y; sh1l[l16 * 3 + 2] = s1z;
    } else {
#pragma unroll
      for (int k = 0; k < 16; ++k) {
        const float v = h1[s * 64 + k];
        tileB[afrag_idx(l16, 48 + k)] = (bf16_t)v;
        xsl[l16 * 16 + k] = v;
      }
    }
    // per-u derived coefficients (each half handles 8 u's)
#pragma unroll
    for (int jj = 0; jj < 8; ++jj) {
      const int u = 8 * half + jj;
      const float x0 = h1[s * 64 + 16 + u * 3 + 0];
      const float x1 = h1[s * 64 + 16 + u * 3 + 1];
      const float x2 = h1[s * 64 + 16 + u * 3 + 2];
      const int o = (l16 * 16 + u) * 3;
      xvl[o + 0] = x0; xvl[o + 1] = x1; xvl[o + 2] = x2;
      d3l[l16 * 16 + u] = x0 * s1x + x1 * s1y + x2 * s1z;
      crl[o + 0] = x1 * s1z - x2 * s1y;
      crl[o + 1] = x2 * s1x - x0 * s1z;
      crl[o + 2] = x0 * s1y - x1 * s1x;
      xvm2l[o + 0] = x0 * m2[0][0] + x1 * m2[1][0] + x2 * m2[2][0];
      xvm2l[o + 1] = x0 * m2[0][1] + x1 * m2[1][1] + x2 * m2[2][1];
      xvm2l[o + 2] = x0 * m2[0][2] + x1 * m2[1][2] + x2 * m2[2][2];
    }
  }

  // GEMM1: hmid = relu(ef @ w1 + b1)
  {
    const v16bf a0 = load_a_packed(tileB, l16, half, 0, false);
    const v16bf a1 = load_a_packed(tileB, l16, half, 1, false);
    float hm[3][8];
#pragma unroll
    for (int nb = 0; nb < 3; ++nb) {
      v8f c = {};
      c = wmma_bf16(a0, load_b_frag(w1T, 64, nb * 16, l16, half, 0, 64), c);
      c = wmma_bf16(a1, load_b_frag(w1T, 64, nb * 16, l16, half, 32, 64), c);
      const float bias = b1s[nb * 16 + l16];
#pragma unroll
      for (int r = 0; r < 8; ++r) hm[nb][r] = fmaxf(c[r] + bias, 0.0f);
    }
#pragma unroll
    for (int nb = 0; nb < 3; ++nb)
#pragma unroll
      for (int r = 0; r < 8; ++r)
        tileB[afrag_idx(r + 8 * half, nb * 16 + l16)] = (bf16_t)hm[nb][r];
  }

  // GEMM2 fused with the 6-matrix contraction; one tight loop per matrix m.
  float S0[8] = {}, S1[8] = {}, S3[8] = {};
  float V2[3][8] = {}, V4[3][8] = {}, V5[3][8] = {};
  {
    const v16bf ha0 = load_a_packed(tileB, l16, half, 0, false);
    const v16bf ha1 = load_a_packed(tileB, l16, half, 1, true);

    auto gemm_block = [&](int n0) -> v8f {
      v8f c = {};
      c = wmma_bf16(ha0, load_b_frag(w2T, 48, n0, l16, half, 0, 48), c);
      c = wmma_bf16(ha1, load_b_frag(w2T, 48, n0, l16, half, 32, 48), c);
      return c;
    };

#pragma unroll 4
    for (int u = 0; u < 16; ++u) {          // m=0: S0 += xs*W
      const v8f c = gemm_block(u * 16);
      const float bias = b2s[u * 16 + l16];
#pragma unroll
      for (int r = 0; r < 8; ++r)
        S0[r] += xsl[(r + 8 * half) * 16 + u] * (c[r] + bias);
    }
#pragma unroll 4
    for (int u = 0; u < 16; ++u) {          // m=1: S1 += xs*W
      const int n0 = 256 + u * 16;
      const v8f c = gemm_block(n0);
      const float bias = b2s[n0 + l16];
#pragma unroll
      for (int r = 0; r < 8; ++r)
        S1[r] += xsl[(r + 8 * half) * 16 + u] * (c[r] + bias);
    }
#pragma unroll 4
    for (int u = 0; u < 16; ++u) {          // m=2: V2 += xv*W
      const int n0 = 512 + u * 16;
      const v8f c = gemm_block(n0);
      const float bias = b2s[n0 + l16];
#pragma unroll
      for (int r = 0; r < 8; ++r) {
        const int o = ((r + 8 * half) * 16 + u) * 3;
        const float val = c[r] + bias;
        V2[0][r] += xvl[o + 0] * val;
        V2[1][r] += xvl[o + 1] * val;
        V2[2][r] += xvl[o + 2] * val;
      }
    }
#pragma unroll 4
    for (int u = 0; u < 16; ++u) {          // m=3: S3 += (xv.sh1)*W
      const int n0 = 768 + u * 16;
      const v8f c = gemm_block(n0);
      const float bias = b2s[n0 + l16];
#pragma unroll
      for (int r = 0; r < 8; ++r)
        S3[r] += d3l[(r + 8 * half) * 16 + u] * (c[r] + bias);
    }
#pragma unroll 4
    for (int u = 0; u < 16; ++u) {          // m=4: V4 += (xv x sh1)*W
      const int n0 = 1024 + u * 16;
      const v8f c = gemm_block(n0);
      const float bias = b2s[n0 + l16];
#pragma unroll
      for (int r = 0; r < 8; ++r) {
        const int o = ((r + 8 * half) * 16 + u) * 3;
        const float val = c[r] + bias;
        V4[0][r] += crl[o + 0] * val;
        V4[1][r] += crl[o + 1] * val;
        V4[2][r] += crl[o + 2] * val;
      }
    }
#pragma unroll 4
    for (int u = 0; u < 16; ++u) {          // m=5: V5 += (xv.m2)*W
      const int n0 = 1280 + u * 16;
      const v8f c = gemm_block(n0);
      const float bias = b2s[n0 + l16];
#pragma unroll
      for (int r = 0; r < 8; ++r) {
        const int o = ((r + 8 * half) * 16 + u) * 3;
        const float val = c[r] + bias;
        V5[0][r] += xvm2l[o + 0] * val;
        V5[1][r] += xvm2l[o + 1] * val;
        V5[2][r] += xvm2l[o + 2] * val;
      }
    }
  }

  // scatter: cols [0..15]=o0, [16+3w+j]=o1o, [64+3w+k]=o1e
#pragma unroll
  for (int r = 0; r < 8; ++r) {
    const int er = r + 8 * half;
    const int d = dstl[er];
    atomicAdd(&up1[d * 112 + l16], C1_P1 * S0[r] + C1_P4 * S3[r]);
#pragma unroll
    for (int j = 0; j < 3; ++j) {
      const float o1o = C1_P2 * S1[r] * sh1l[er * 3 + j] + C1_P3 * V2[j][r] +
                        C1_P6 * V5[j][r];
      atomicAdd(&up1[d * 112 + 16 + 3 * l16 + j], o1o);
      atomicAdd(&up1[d * 112 + 64 + 3 * l16 + j], C1_P5 * V4[j][r]);
    }
  }
}

// ---------------------------------------------------------------- edge head
__global__ void edge_head_kernel(const float* __restrict__ pos,
                                 const int* __restrict__ src, const int* __restrict__ dst,
                                 const float* __restrict__ h2, const int* __restrict__ batch,
                                 const float* __restrict__ w1, const float* __restrict__ b1,
                                 const float* __restrict__ w2, const float* __restrict__ b2,
                                 const float* __restrict__ w3, const float* __restrict__ b3,
                                 float* __restrict__ gsum, float* __restrict__ gcnt) {
  const int e = blockIdx.x * blockDim.x + threadIdx.x;
  if (e >= N_EDGES) return;
  const int s = src[e], d = dst[e];
  float ein[64];
  {
    const float dx = pos[d * 3 + 0] - pos[s * 3 + 0];
    const float dy = pos[d * 3 + 1] - pos[s * 3 + 1];
    const float dz = pos[d * 3 + 2] - pos[s * 3 + 2];
    const float rr = sqrtf(dx * dx + dy * dy + dz * dz);
#pragma unroll
    for (int k = 0; k < 32; ++k) {
      const float t = rr - DMU * (float)k;
      ein[k] = __expf(RBFC * t * t);
    }
  }
#pragma unroll
  for (int k = 0; k < 16; ++k) {
    ein[32 + k] = h2[s * 112 + k];  // e_in = [edge_attr, h[src], h[dst]]
    ein[48 + k] = h2[d * 112 + k];
  }
  float hA[16];
#pragma unroll
  for (int j = 0; j < 16; ++j) hA[j] = b1[j];
  for (int k = 0; k < 64; ++k) {
    const float v = ein[k];
#pragma unroll
    for (int j = 0; j < 16; ++j) hA[j] += v * w1[k * 16 + j];
  }
#pragma unroll
  for (int j = 0; j < 16; ++j) hA[j] = fmaxf(hA[j], 0.0f);
  float hB[16];
#pragma unroll
  for (int j = 0; j < 16; ++j) hB[j] = b2[j];
#pragma unroll
  for (int k = 0; k < 16; ++k)
#pragma unroll
    for (int j = 0; j < 16; ++j) hB[j] += hA[k] * w2[k * 16 + j];
  float o = b3[0];
#pragma unroll
  for (int k = 0; k < 16; ++k) o += fmaxf(hB[k], 0.0f) * w3[k];
  const int g = batch[s];
  atomicAdd(&gsum[g], o);
  atomicAdd(&gcnt[g], 1.0f);
}

// ---------------------------------------------------------------- node head
__global__ void node_head_kernel(const float* __restrict__ h2, const int* __restrict__ batch,
                                 const float* __restrict__ w1, const float* __restrict__ b1,
                                 const float* __restrict__ w2, const float* __restrict__ b2,
                                 const float* __restrict__ w3, const float* __restrict__ b3,
                                 float* __restrict__ gsum, float* __restrict__ gcnt) {
  const int n = blockIdx.x * blockDim.x + threadIdx.x;
  if (n >= N_NODES) return;
  float hA[32];
#pragma unroll
  for (int j = 0; j < 32; ++j) hA[j] = b1[j];
#pragma unroll
  for (int k = 0; k < 16; ++k) {
    const float v = h2[n * 112 + k];
#pragma unroll
    for (int j = 0; j < 32; ++j) hA[j] += v * w1[k * 32 + j];
  }
#pragma unroll
  for (int j = 0; j < 32; ++j) hA[j] = fmaxf(hA[j], 0.0f);
  float hB[16];
#pragma unroll
  for (int j = 0; j < 16; ++j) hB[j] = b2[j];
#pragma unroll
  for (int k = 0; k < 32; ++k)
#pragma unroll
    for (int j = 0; j < 16; ++j) hB[j] += hA[k] * w2[k * 16 + j];
  float o = b3[0];
#pragma unroll
  for (int k = 0; k < 16; ++k) o += fmaxf(hB[k], 0.0f) * w3[k];
  const int g = batch[n];
  atomicAdd(&gsum[g], o);
  atomicAdd(&gcnt[g], 1.0f);
}

// ---------------------------------------------------------------- finalize
__global__ void finalize_kernel(const float* __restrict__ gacc, float* __restrict__ out) {
  const int g = threadIdx.x;
  if (g < N_GRAPHS)
    out[g] = gacc[g] / fmaxf(gacc[8 + g], 1.0f) +
             gacc[16 + g] / fmaxf(gacc[24 + g], 1.0f);
}

// ================================================================ launch
extern "C" void kernel_launch(void* const* d_in, const int* in_sizes, int n_in,
                              void* d_out, int out_size, void* d_ws, size_t ws_size,
                              hipStream_t stream) {
  (void)in_sizes; (void)n_in; (void)out_size; (void)ws_size;
  const float* x     = (const float*)d_in[0];
  const float* pos   = (const float*)d_in[1];
  const int*   eidx  = (const int*)d_in[2];
  const int*   batch = (const int*)d_in[3];
  const int* srcp = eidx;
  const int* dstp = eidx + N_EDGES;
  // p_emb 4..7, p_conv0 8..11, p_conv1 12..15, p_edge_head 16..21, p_node_head 22..27
  const float* pe_w1 = (const float*)d_in[4];
  const float* pe_b1 = (const float*)d_in[5];
  const float* pe_w2 = (const float*)d_in[6];
  const float* pe_b2 = (const float*)d_in[7];
  const float* c0_w1 = (const float*)d_in[8];
  const float* c0_b1 = (const float*)d_in[9];
  const float* c0_w2 = (const float*)d_in[10];
  const float* c0_b2 = (const float*)d_in[11];
  const float* c1_w1 = (const float*)d_in[12];
  const float* c1_b1 = (const float*)d_in[13];
  const float* c1_w2 = (const float*)d_in[14];
  const float* c1_b2 = (const float*)d_in[15];
  const float* eh_w1 = (const float*)d_in[16];
  const float* eh_b1 = (const float*)d_in[17];
  const float* eh_w2 = (const float*)d_in[18];
  const float* eh_b2 = (const float*)d_in[19];
  const float* eh_w3 = (const float*)d_in[20];
  const float* eh_b3 = (const float*)d_in[21];
  const float* nh_w1 = (const float*)d_in[22];
  const float* nh_b1 = (const float*)d_in[23];
  const float* nh_w2 = (const float*)d_in[24];
  const float* nh_b2 = (const float*)d_in[25];
  const float* nh_w3 = (const float*)d_in[26];
  const float* nh_b3 = (const float*)d_in[27];

  float* ws   = (float*)d_ws;
  float* up0  = ws;                          // 8192*64
  float* up1  = up0 + N_NODES * 64;          // 8192*112
  float* deg  = up1 + N_NODES * 112;         // 8192
  float* gacc = deg + N_NODES;               // 32: se_sum, se_cnt, sn_sum, sn_cnt
  float* h0 = gacc + 32;                     // 8192*16
  float* h1 = h0 + N_NODES * 16;             // 8192*64
  float* h2 = h1 + N_NODES * 64;             // 8192*112
  const size_t zero_floats = (size_t)N_NODES * 64 + (size_t)N_NODES * 112 + N_NODES + 32;
  hipMemsetAsync(d_ws, 0, zero_floats * sizeof(float), stream);

  embed_kernel<<<N_NODES / 256, 256, 0, stream>>>(x, pe_w1, pe_b1, pe_w2, pe_b2, h0);
  deg_kernel<<<N_EDGES / 256, 256, 0, stream>>>(dstp, deg);

  const size_t LDS0 = 48 * 64 * 2 + 48 * 4 + 512 * 48 * 2 + 512 * 4 + 8 * 3328;    // 84160
  conv0_kernel<<<N_EDGES / 128, 256, LDS0, stream>>>(pos, srcp, dstp, h0,
                                                     c0_w1, c0_b1, c0_w2, c0_b2, up0);
  node_update_kernel<<<(N_NODES * 64) / 256, 256, 0, stream>>>(h0, 16, up0, 64, deg, h1);

  const size_t LDS1 = 48 * 64 * 2 + 48 * 4 + 1536 * 48 * 2 + 1536 * 4 + 8 * 13568; // 268480
  conv1_kernel<<<N_EDGES / 128, 256, LDS1, stream>>>(pos, srcp, dstp, h1,
                                                     c1_w1, c1_b1, c1_w2, c1_b2, up1);
  node_update_kernel<<<(N_NODES * 112) / 256, 256, 0, stream>>>(h1, 64, up1, 112, deg, h2);

  edge_head_kernel<<<N_EDGES / 256, 256, 0, stream>>>(pos, srcp, dstp, h2, batch,
                                                      eh_w1, eh_b1, eh_w2, eh_b2, eh_w3, eh_b3,
                                                      gacc, gacc + 8);
  node_head_kernel<<<N_NODES / 256, 256, 0, stream>>>(h2, batch,
                                                      nh_w1, nh_b1, nh_w2, nh_b2, nh_w3, nh_b3,
                                                      gacc + 16, gacc + 24);
  finalize_kernel<<<1, 32, 0, stream>>>(gacc, (float*)d_out);
}
